// ToMe16_mlp_31181462569519
// MI455X (gfx1250) — matrix operand order repaired and checked by reading, not compiled, not run on hardware
//
#include <hip/hip_runtime.h>
#include <hip/hip_bf16.h>

// ---------------------------------------------------------------------------
// ToMe-16 + MLP for MI455X (gfx1250, wave32, WMMA bf16 16x16x32,
// async global->LDS staging, fused scores+argmax)
// ---------------------------------------------------------------------------

typedef __attribute__((ext_vector_type(16))) __bf16 bf16x16;
typedef __attribute__((ext_vector_type(8)))  float  f32x8;

struct Frag { uint4 q[2]; };   // 32 bytes == 16 bf16 == one WMMA A/B fragment per lane

__device__ __forceinline__ f32x8 wmma_bf16(const Frag& a, const Frag& b, f32x8 c) {
  return __builtin_amdgcn_wmma_f32_16x16x32_bf16(
      false, __builtin_bit_cast(bf16x16, a),
      false, __builtin_bit_cast(bf16x16, b),
      (short)0, c, false, false);
}

__device__ __forceinline__ unsigned short f2bf(float f) {
  unsigned int u = __float_as_uint(f);
  u += 0x7FFFu + ((u >> 16) & 1u);          // round-to-nearest-even
  return (unsigned short)(u >> 16);
}

// LDS aperture: generic shared pointer low 32 bits == wave-relative LDS offset
__device__ __forceinline__ unsigned lds_off(const void* p) {
  return (unsigned)(unsigned long long)p;
}

// CDNA5 async DMA: global -> LDS, 16 bytes/lane, tracked by ASYNCcnt
__device__ __forceinline__ void async_to_lds_b128(unsigned ldsByteOff, const void* gaddr) {
  asm volatile("global_load_async_to_lds_b128 %0, %1, off"
               :: "v"(ldsByteOff), "v"(gaddr) : "memory");
}
__device__ __forceinline__ void wait_asynccnt0() {
  asm volatile("s_wait_asynccnt 0x0" ::: "memory");
}

#define NB     8
#define C_DIM  1024
#define H_DIM  4096
#define P0     4608

// ---------------------------------------------------------------------------
// 1) x + pos_embed  (broadcast over batch)
// ---------------------------------------------------------------------------
__global__ void addpos_kernel(const float* __restrict__ x,
                              const float* __restrict__ pos,
                              float* __restrict__ xout) {
  size_t i = (size_t)blockIdx.x * blockDim.x + threadIdx.x;
  size_t total = (size_t)NB * P0 * C_DIM;
  if (i >= total) return;
  size_t pc = i % ((size_t)P0 * C_DIM);
  xout[i] = x[i] + pos[pc];
}

__global__ void fill_ones_kernel(float* __restrict__ p, int n) {
  int i = blockIdx.x * blockDim.x + threadIdx.x;
  if (i < n) p[i] = 1.0f;
}

// ---------------------------------------------------------------------------
// 2) metric: head-mean (16 heads x 64) + L2 normalize, de-interleave even/odd
// ---------------------------------------------------------------------------
__global__ __launch_bounds__(64)
void metric_kernel(const float* __restrict__ x,          // [B][pp][1024]
                   unsigned short* __restrict__ mA,      // [B][pp/2][64] even tokens
                   unsigned short* __restrict__ mB,      // [B][pp/2][64] odd tokens
                   int pp) {
  int p = blockIdx.x, b = blockIdx.y, j = threadIdx.x;
  const float* xr = x + ((size_t)b * pp + p) * C_DIM;
  float s = 0.f;
#pragma unroll
  for (int h = 0; h < 16; ++h) s += xr[h * 64 + j];
  s *= (1.0f / 16.0f);
  __shared__ float red[64];
  red[j] = s * s;
  __syncthreads();
  for (int off = 32; off > 0; off >>= 1) {
    if (j < off) red[j] += red[j + off];
    __syncthreads();
  }
  float m = s * rsqrtf(red[0]);
  unsigned short bf = f2bf(m);
  int n = pp >> 1;
  size_t o = ((size_t)b * n + (p >> 1)) * 64 + j;
  if (p & 1) mB[o] = bf; else mA[o] = bf;
}

// ---------------------------------------------------------------------------
// 3) scores = a_m @ b_m^T fused with row max/argmax; software-pipelined
//    B-fragment loads so global latency overlaps the WMMAs.
//    C layout: lane = r + 16*half holds N = r, M = 8*half + v.
// ---------------------------------------------------------------------------
__global__ __launch_bounds__(128)
void scores_kernel(const unsigned short* __restrict__ mA,   // [B][n][64]
                   const unsigned short* __restrict__ mB,   // [B][n][64]
                   float* __restrict__ nodeMax,             // [B][n]
                   int*   __restrict__ nodeIdx,             // [B][n]
                   int n) {
  int b = blockIdx.y;
  int warp = threadIdx.x >> 5;
  int nTiles = (n + 15) >> 4;
  int rowTile = blockIdx.x * 4 + warp;
  if (rowTile >= nTiles) return;                  // wave-uniform
  int lane = threadIdx.x & 31;
  int half = lane >> 4, r = lane & 15;

  const unsigned short* Ab = mA + (size_t)b * n * 64;
  const unsigned short* Bb = mB + (size_t)b * n * 64;

  int rowA = rowTile * 16 + r;                    // A lane row (clamped load)
  if (rowA >= n) rowA = n - 1;
  const unsigned short* ar = Ab + (size_t)rowA * 64;
  Frag a0, a1;
  a0.q[0] = *(const uint4*)(ar +  0 + half * 16);  // K 0..7 / 16..23
  a0.q[1] = *(const uint4*)(ar +  8 + half * 16);  // K 8..15 / 24..31
  a1.q[0] = *(const uint4*)(ar + 32 + half * 16);
  a1.q[1] = *(const uint4*)(ar + 40 + half * 16);

  float best[8];
  int   bidx[8];
#pragma unroll
  for (int v = 0; v < 8; ++v) { best[v] = -3.4e38f; bidx[v] = 0; }

  // pipelined B fragment loads
  Frag b0n, b1n;
  {
    int colc = (r < n) ? r : n - 1;
    const unsigned short* br = Bb + (size_t)colc * 64;
    b0n.q[0] = *(const uint4*)(br +  0 + half * 16);
    b0n.q[1] = *(const uint4*)(br +  8 + half * 16);
    b1n.q[0] = *(const uint4*)(br + 32 + half * 16);
    b1n.q[1] = *(const uint4*)(br + 40 + half * 16);
  }

  for (int ct = 0; ct < nTiles; ++ct) {
    Frag b0 = b0n, b1 = b1n;
    if (ct + 1 < nTiles) {
      int coln = (ct + 1) * 16 + r;
      int colc = coln < n ? coln : n - 1;
      const unsigned short* br = Bb + (size_t)colc * 64;
      b0n.q[0] = *(const uint4*)(br +  0 + half * 16);
      b0n.q[1] = *(const uint4*)(br +  8 + half * 16);
      b1n.q[0] = *(const uint4*)(br + 32 + half * 16);
      b1n.q[1] = *(const uint4*)(br + 40 + half * 16);
    }

    f32x8 c = {0.f,0.f,0.f,0.f,0.f,0.f,0.f,0.f};
    c = wmma_bf16(a0, b0, c);
    c = wmma_bf16(a1, b1, c);

    int col = ct * 16 + r;
    bool valid = (col < n);
#pragma unroll
    for (int v = 0; v < 8; ++v) {
      float s = c[v];
      if (valid && s > best[v]) { best[v] = s; bidx[v] = col; }
    }
  }
  // reduce over the 16 lane-columns inside each half (argmax, first-index ties)
#pragma unroll
  for (int mask = 1; mask <= 8; mask <<= 1) {
#pragma unroll
    for (int v = 0; v < 8; ++v) {
      float om = __shfl_xor(best[v], mask, 32);
      int   oi = __shfl_xor(bidx[v], mask, 32);
      if (om > best[v] || (om == best[v] && oi < bidx[v])) { best[v] = om; bidx[v] = oi; }
    }
  }
  if (r == 0) {
#pragma unroll
    for (int v = 0; v < 8; ++v) {
      int row = rowTile * 16 + half * 8 + v;
      if (row < n) {
        nodeMax[(size_t)b * n + row] = best[v];
        nodeIdx[(size_t)b * n + row] = bidx[v];
      }
    }
  }
}

// ---------------------------------------------------------------------------
// 4) per-batch bitonic argsort of -nodeMax (ascending, stable via index tie)
// ---------------------------------------------------------------------------
__global__ __launch_bounds__(1024)
void sort_kernel(const float* __restrict__ nodeMax,
                 int* __restrict__ edgeIdx, int n, int SZ) {
  __shared__ float key[4096];
  __shared__ int   idx[4096];
  int b = blockIdx.x, tid = threadIdx.x, nt = blockDim.x;
  for (int i = tid; i < SZ; i += nt) {
    key[i] = (i < n) ? -nodeMax[(size_t)b * n + i] : 3.4e38f;
    idx[i] = i;
  }
  __syncthreads();
  for (int k = 2; k <= SZ; k <<= 1) {
    for (int j = k >> 1; j > 0; j >>= 1) {
      for (int i = tid; i < SZ; i += nt) {
        int ixj = i ^ j;
        if (ixj > i) {
          float k1 = key[i], k2 = key[ixj];
          int   i1 = idx[i], i2 = idx[ixj];
          bool up = ((i & k) == 0);
          bool gt = (k1 > k2) || (k1 == k2 && i1 > i2);
          if (gt == up) { key[i] = k2; key[ixj] = k1; idx[i] = i2; idx[ixj] = i1; }
        }
      }
      __syncthreads();
    }
  }
  for (int i = tid; i < n; i += nt) edgeIdx[(size_t)b * n + i] = idx[i];
}

// ---------------------------------------------------------------------------
// 5) merge: gather unm+dst (x*size), scatter-add src, divide by merged size
// ---------------------------------------------------------------------------
__global__ __launch_bounds__(256)
void merge_init_kernel(const float* __restrict__ xin, const float* __restrict__ szin,
                       const int* __restrict__ edge,
                       float* __restrict__ xout, float* __restrict__ szout,
                       int n, int rr, int tokOut, int ppIn) {
  int t = blockIdx.x, b = blockIdx.y;
  int unmCount = n - rr;
  int srcTok;
  if (t < unmCount) srcTok = 2 * edge[(size_t)b * n + rr + t];   // unm (even tokens)
  else              srcTok = 2 * (t - unmCount) + 1;             // dst (odd tokens)
  const float* xs = xin + ((size_t)b * ppIn + srcTok) * C_DIM;
  float sz = szin[(size_t)b * ppIn + srcTok];
  float* xd = xout + ((size_t)b * tokOut + t) * C_DIM;
  for (int c = threadIdx.x; c < C_DIM; c += blockDim.x) xd[c] = xs[c] * sz;
  if (threadIdx.x == 0) szout[(size_t)b * tokOut + t] = sz;
}

__global__ __launch_bounds__(256)
void merge_scatter_kernel(const float* __restrict__ xin, const float* __restrict__ szin,
                          const int* __restrict__ edge, const int* __restrict__ nodeIdx,
                          float* __restrict__ xout, float* __restrict__ szout,
                          int n, int rr, int tokOut, int ppIn) {
  int k = blockIdx.x, b = blockIdx.y;          // k < rr
  int s = edge[(size_t)b * n + k];
  int srcTok = 2 * s;
  int d = nodeIdx[(size_t)b * n + s];
  int unmCount = n - rr;
  const float* xs = xin + ((size_t)b * ppIn + srcTok) * C_DIM;
  float sz = szin[(size_t)b * ppIn + srcTok];
  float* xd = xout + ((size_t)b * tokOut + unmCount + d) * C_DIM;
  for (int c = threadIdx.x; c < C_DIM; c += blockDim.x) atomicAdd(&xd[c], xs[c] * sz);
  if (threadIdx.x == 0) atomicAdd(&szout[(size_t)b * tokOut + unmCount + d], sz);
}

__global__ __launch_bounds__(256)
void merge_div_kernel(float* __restrict__ xout, const float* __restrict__ szout,
                      int tokOut) {
  int t = blockIdx.x, b = blockIdx.y;
  float sz = szout[(size_t)b * tokOut + t];
  float* xd = xout + ((size_t)b * tokOut + t) * C_DIM;
  for (int c = threadIdx.x; c < C_DIM; c += blockDim.x) xd[c] = xd[c] / sz;
}

// ---------------------------------------------------------------------------
// 6) fp32 -> bf16 conversion
// ---------------------------------------------------------------------------
__global__ void cvt_bf16_kernel(const float* __restrict__ src,
                                unsigned short* __restrict__ dst, size_t n) {
  size_t i = (size_t)blockIdx.x * blockDim.x + threadIdx.x;
  if (i < n) dst[i] = f2bf(src[i]);
}

// ---------------------------------------------------------------------------
// 7) WMMA GEMM: C = A(MxK,bf16) @ B(KxN,bf16) + bias; epilogue GELU->bf16 or
//    plain->fp32. Block tile 128x128, 8 waves (2x4), wave tile 64x32.
//    A tile staged via async global->LDS DMA (ASYNCcnt); B tile transposed
//    through VGPRs into LDS so fragment reads are contiguous 32B ds loads.
// ---------------------------------------------------------------------------
#define BM 128
#define BN 128
#define BK 32

__global__ __launch_bounds__(256)
void gemm_bf16_kernel(const unsigned short* __restrict__ A,
                      const unsigned short* __restrict__ Bm,
                      const float* __restrict__ bias,
                      unsigned short* __restrict__ outBf,
                      float* __restrict__ outF,
                      int M, int N, int K, int mode) {
  __shared__ unsigned short ldsA[BM * BK];     // [row][k]
  __shared__ unsigned short ldsBt[BN * BK];    // [col][k]  (transposed)
  int tid = threadIdx.x;
  int lane = tid & 31, warp = tid >> 5;
  int warpM = warp & 1, warpN = warp >> 1;     // 2 x 4 wave grid
  int half = lane >> 4, r = lane & 15;
  int mBlock = blockIdx.y * BM, nBlock = blockIdx.x * BN;

  f32x8 acc[4][2];
#pragma unroll
  for (int mf = 0; mf < 4; ++mf)
#pragma unroll
    for (int nf = 0; nf < 2; ++nf)
      acc[mf][nf] = (f32x8){0.f,0.f,0.f,0.f,0.f,0.f,0.f,0.f};

  int aRow = tid >> 1, aKp = (tid & 1) * 16;   // A staging: 32B/thread (2 x b128)
  int bK = tid >> 3, bN0 = (tid & 7) * 16;     // B staging: 16 bf16/thread
  unsigned aLds0 = lds_off(ldsA + aRow * BK + aKp);
  unsigned aLds1 = aLds0 + 16;

  for (int k0 = 0; k0 < K; k0 += BK) {
    // A tile: async DMA global -> LDS (no VGPR round trip)
    {
      const char* g = (const char*)(A + (size_t)(mBlock + aRow) * K + k0 + aKp);
      async_to_lds_b128(aLds0, g);
      async_to_lds_b128(aLds1, g + 16);
    }
    // B tile: load 16 bf16 along N, store transposed [col][k]
    {
      const unsigned short* src = Bm + (size_t)(k0 + bK) * N + nBlock + bN0;
#pragma unroll
      for (int i = 0; i < 16; ++i) ldsBt[(bN0 + i) * BK + bK] = src[i];
    }
    wait_asynccnt0();       // my async writes to LDS are done
    __syncthreads();        // everyone's writes are done

    if (k0 + BK < K) {      // prefetch next B K-tile -> global_prefetch_b8
      __builtin_prefetch(Bm + (size_t)(k0 + BK + bK) * N + nBlock + bN0, 0, 1);
    }

    Frag af[4], bf[2];
#pragma unroll
    for (int mf = 0; mf < 4; ++mf) {
      int row = warpM * 64 + mf * 16 + r;
      af[mf].q[0] = *(const uint4*)(ldsA + row * BK + half * 16);
      af[mf].q[1] = *(const uint4*)(ldsA + row * BK + 8 + half * 16);
    }
#pragma unroll
    for (int nf = 0; nf < 2; ++nf) {
      int col = warpN * 32 + nf * 16 + r;
      const uint4* p = (const uint4*)(ldsBt + col * BK + half * 16);
      bf[nf].q[0] = p[0];
      bf[nf].q[1] = p[1];
    }
#pragma unroll
    for (int mf = 0; mf < 4; ++mf)
#pragma unroll
      for (int nf = 0; nf < 2; ++nf)
        acc[mf][nf] = wmma_bf16(af[mf], bf[nf], acc[mf][nf]);

    __syncthreads();
  }

#pragma unroll
  for (int mf = 0; mf < 4; ++mf) {
#pragma unroll
    for (int nf = 0; nf < 2; ++nf) {
      int colBase = nBlock + warpN * 32 + nf * 16 + r;
      float bv = bias[colBase];
#pragma unroll
      for (int v = 0; v < 8; ++v) {
        int row = mBlock + warpM * 64 + mf * 16 + half * 8 + v;
        float x = acc[mf][nf][v] + bv;
        if (mode == 0) {
          float g = 0.5f * x * (1.0f + erff(x * 0.70710678118654752f));
          outBf[(size_t)row * N + colBase] = f2bf(g);
        } else {
          outF[(size_t)row * N + colBase] = x;
        }
      }
    }
  }
}

// ---------------------------------------------------------------------------
// Host orchestration
// ---------------------------------------------------------------------------
extern "C" void kernel_launch(void* const* d_in, const int* in_sizes, int n_in,
                              void* d_out, int out_size, void* d_ws, size_t ws_size,
                              hipStream_t stream) {
  (void)in_sizes; (void)n_in; (void)out_size; (void)ws_size;
  const float* x   = (const float*)d_in[0];
  const float* pos = (const float*)d_in[1];
  const float* W1  = (const float*)d_in[2];
  const float* b1  = (const float*)d_in[3];
  const float* W2  = (const float*)d_in[4];
  const float* b2  = (const float*)d_in[5];

  char* w = (char*)d_ws;
  size_t off = 0;
  auto alloc = [&](size_t bytes) -> void* {
    off = (off + 255) & ~(size_t)255;
    void* p = w + off;
    off += bytes;
    return p;
  };
  float* X0 = (float*)alloc((size_t)NB * P0 * C_DIM * 4);       // 151 MB
  float* X1 = (float*)alloc((size_t)NB * 2304 * C_DIM * 4);     //  75 MB
  float* S0 = (float*)alloc((size_t)NB * P0 * 4);
  float* S1 = (float*)alloc((size_t)NB * 2304 * 4);
  unsigned short* mA = (unsigned short*)alloc((size_t)NB * 2304 * 64 * 2);
  unsigned short* mB = (unsigned short*)alloc((size_t)NB * 2304 * 64 * 2);
  float* nodeMax = (float*)alloc((size_t)NB * 2304 * 4);
  int*   nodeIdx = (int*)alloc((size_t)NB * 2304 * 4);
  int*   edgeIdx = (int*)alloc((size_t)NB * 2304 * 4);
  unsigned short* Xb  = (unsigned short*)alloc((size_t)1024 * C_DIM * 2);
  unsigned short* W1b = (unsigned short*)alloc((size_t)C_DIM * H_DIM * 2);
  unsigned short* W2b = (unsigned short*)alloc((size_t)H_DIM * H_DIM * 2);
  unsigned short* Hb  = (unsigned short*)alloc((size_t)1024 * H_DIM * 2);

  // 1) x + pos_embed, size = 1
  {
    size_t total = (size_t)NB * P0 * C_DIM;
    addpos_kernel<<<dim3((unsigned)((total + 255) / 256)), 256, 0, stream>>>(x, pos, X0);
    int ntok = NB * P0;
    fill_ones_kernel<<<dim3((ntok + 255) / 256), 256, 0, stream>>>(S0, ntok);
  }

  // 2) ToMe rounds: pp 4608->2304->1152->576->288->144->128
  const int rs[6] = {2304, 1152, 576, 288, 144, 16};
  int pp = P0;
  float *xin = X0, *xout = X1, *szin = S0, *szout = S1;
  for (int ri = 0; ri < 6; ++ri) {
    int rr = rs[ri];
    int n = pp >> 1;
    int unmCount = n - rr;
    int tokOut = unmCount + n;

    metric_kernel<<<dim3(pp, NB), 64, 0, stream>>>(xin, mA, mB, pp);

    int nTiles = (n + 15) >> 4;
    scores_kernel<<<dim3((nTiles + 3) / 4, NB), 128, 0, stream>>>(mA, mB, nodeMax, nodeIdx, n);

    int SZ = 1;
    while (SZ < n) SZ <<= 1;
    sort_kernel<<<dim3(NB), 1024, 0, stream>>>(nodeMax, edgeIdx, n, SZ);

    merge_init_kernel<<<dim3(tokOut, NB), 256, 0, stream>>>(
        xin, szin, edgeIdx, xout, szout, n, rr, tokOut, pp);
    merge_scatter_kernel<<<dim3(rr, NB), 256, 0, stream>>>(
        xin, szin, edgeIdx, nodeIdx, xout, szout, n, rr, tokOut, pp);
    merge_div_kernel<<<dim3(tokOut, NB), 256, 0, stream>>>(xout, szout, tokOut);

    pp = tokOut;
    float* t;
    t = xin;  xin = xout;  xout = t;
    t = szin; szin = szout; szout = t;
  }
  // xin now holds [8][128][1024] (pp == 128)

  // 3) convert operands to bf16
  {
    size_t nXb = (size_t)1024 * C_DIM;
    cvt_bf16_kernel<<<dim3((unsigned)((nXb + 255) / 256)), 256, 0, stream>>>(xin, Xb, nXb);
    size_t nW1 = (size_t)C_DIM * H_DIM;
    cvt_bf16_kernel<<<dim3((unsigned)((nW1 + 255) / 256)), 256, 0, stream>>>(W1, W1b, nW1);
    size_t nW2 = (size_t)H_DIM * H_DIM;
    cvt_bf16_kernel<<<dim3((unsigned)((nW2 + 255) / 256)), 256, 0, stream>>>(W2, W2b, nW2);
  }

  // 4) MLP: Hb = gelu(Xb@W1 + b1)  ->  out = Hb@W2 + b2
  gemm_bf16_kernel<<<dim3(H_DIM / BN, 1024 / BM), 256, 0, stream>>>(
      Xb, W1b, b1, Hb, nullptr, 1024, H_DIM, C_DIM, 0);
  gemm_bf16_kernel<<<dim3(H_DIM / BN, 1024 / BM), 256, 0, stream>>>(
      Hb, W2b, b2, nullptr, (float*)d_out, 1024, H_DIM, H_DIM, 1);
}
